// TransformerBlock_67662914781594
// MI455X (gfx1250) — matrix operand order repaired and checked by reading, not compiled, not run on hardware
//
#include <hip/hip_runtime.h>
#include <hip/hip_bf16.h>

// ---------------------------------------------------------------------------
// CDNA5 (gfx1250) bf16-WMMA transformer block.
// - All matmuls: v_wmma_f32_16x16x32_bf16, fp32 accumulate.
// - GEMM K-loop: double-buffered LDS; A tiles via async global->LDS DMA
//   (global_load_async_to_lds_b128 / s_wait_asynccnt), B tiles via
//   non-temporal VGPR prefetch + fp32->bf16 convert + vector ds stores.
// Workspace requirement: ~219 MB.
// ---------------------------------------------------------------------------

typedef __bf16 bf16_t;
typedef bf16_t v16bf __attribute__((ext_vector_type(16)));
typedef bf16_t v8bf  __attribute__((ext_vector_type(8)));
typedef float  v8f   __attribute__((ext_vector_type(8)));

#define B_    4
#define S_    1024
#define DIM_  4096
#define NH_   32
#define NKVH_ 8
#define HD_   128
#define NREP_ 4
#define HID_  11008
#define M_    (B_ * S_)          // 4096 tokens
#define KVD_  (NKVH_ * HD_)      // 1024

__device__ __forceinline__ v8f wmma_bf16(v16bf a, v16bf b, v8f c) {
  // (neg_a, A, neg_b, B, c_mod, C, reuse_a, reuse_b)
  return __builtin_amdgcn_wmma_f32_16x16x32_bf16(false, a, false, b,
                                                 (short)0, c, false, false);
}

__device__ __forceinline__ v16bf combine8(v8bf lo, v8bf hi) {
  v16bf r;
#pragma unroll
  for (int i = 0; i < 8; ++i) { r[i] = lo[i]; r[8 + i] = hi[i]; }
  return r;
}

// Async global->LDS copy, 16 bytes per lane (tracked on ASYNCcnt).
// Generic LDS addresses carry the byte offset in the low 32 bits.
__device__ __forceinline__ void async_copy_b128(void* lds_ptr, const void* g) {
  uint32_t lds_off = (uint32_t)(uintptr_t)lds_ptr;
  asm volatile("global_load_async_to_lds_b128 %0, %1, off"
               :: "v"(lds_off), "v"(g) : "memory");
}
__device__ __forceinline__ void wait_async0() {
  asm volatile("s_wait_asynccnt 0x0" ::: "memory");
}

// ---------------------------------------------------------------------------
// RMSNorm: one block per row, fp32 in -> bf16 out
// ---------------------------------------------------------------------------
__global__ __launch_bounds__(256)
void rmsnorm_kernel(const float* __restrict__ X, const float* __restrict__ W,
                    bf16_t* __restrict__ O, int ncols) {
  __shared__ float red[256];
  const int row = blockIdx.x;
  const size_t base = (size_t)row * ncols;
  float ss = 0.f;
  for (int j = threadIdx.x; j < ncols; j += 256) {
    float v = X[base + j];
    ss += v * v;
  }
  red[threadIdx.x] = ss;
  __syncthreads();
  for (int s = 128; s > 0; s >>= 1) {
    if (threadIdx.x < s) red[threadIdx.x] += red[threadIdx.x + s];
    __syncthreads();
  }
  const float inv = rsqrtf(red[0] / (float)ncols + 1e-5f);
  for (int j = threadIdx.x; j < ncols; j += 256)
    O[base + j] = (bf16_t)(X[base + j] * inv * W[j]);
}

// ---------------------------------------------------------------------------
// RoPE + cast to bf16.  X: [B,S,nheads,HD] fp32; FC: [S, HD/2] complex64
// (interleaved float pairs).  One thread per (token, head, d/2).
// ---------------------------------------------------------------------------
__global__ __launch_bounds__(256)
void rope_cast_kernel(const float* __restrict__ X, const float* __restrict__ FC,
                      bf16_t* __restrict__ O, int nheads, size_t total) {
  size_t i = (size_t)blockIdx.x * blockDim.x + threadIdx.x;
  if (i >= total) return;
  const int d2 = (int)(i & 63);            // HD/2 = 64
  const size_t t = i >> 6;                 // (token, head)
  const size_t tok = t / nheads;
  const int s = (int)(tok & (S_ - 1));
  const float x0 = X[2 * i], x1 = X[2 * i + 1];
  const float c  = FC[((size_t)s * 64 + d2) * 2];
  const float sn = FC[((size_t)s * 64 + d2) * 2 + 1];
  O[2 * i]     = (bf16_t)(x0 * c - x1 * sn);
  O[2 * i + 1] = (bf16_t)(x0 * sn + x1 * c);
}

__global__ __launch_bounds__(256)
void cast_bf16_kernel(const float* __restrict__ X, bf16_t* __restrict__ O,
                      size_t n) {
  size_t i = (size_t)blockIdx.x * blockDim.x + threadIdx.x;
  if (i < n) O[i] = (bf16_t)X[i];
}

// ---------------------------------------------------------------------------
// Generic GEMM: C[M,N](f32) = A[M,K](bf16) @ B[K,N](f32->bf16) [+ residual]
// Block = 256 threads (8 waves), tile 128x128, K-step 32, double-buffered.
// A tile: async DMA global->LDS.  B tile: NT loads -> regs -> bf16 -> LDS
// (each thread owns a contiguous 16-wide k-run -> two ds_store_b128).
// Wave grid 4x2; each wave computes 32x64 (2x4 WMMA 16x16 tiles).
// ---------------------------------------------------------------------------
__global__ __launch_bounds__(256)
void gemm_bf16_kernel(const bf16_t* __restrict__ A, const float* __restrict__ Bw,
                      float* C, const float* R, int N, int K) {
  __shared__ bf16_t sA[2][128][32];   // 16 KB, row-major
  __shared__ bf16_t sB[2][128][32];   // 16 KB, transposed: [n][k]
  const int tid  = threadIdx.x;
  const int lane = tid & 31, wave = tid >> 5;
  const int half = lane >> 4, l16 = lane & 15;
  const int wm = wave >> 1, wn = wave & 1;
  const int bm0 = blockIdx.y * 128, bn0 = blockIdx.x * 128;

  // Per-thread staging coordinates
  const int ar0 = tid >> 2, ac0 = (tid & 3) << 3;          // A: 2 chunks of 8
  const int ar1 = (tid + 256) >> 2, ac1 = ((tid + 256) & 3) << 3;
  const int bn_ = tid & 127;            // B: column owned by this thread
  const int bk0 = (tid >> 7) << 4;      // contiguous k-run start (0 or 16)

  v8f acc[2][4];
#pragma unroll
  for (int i = 0; i < 2; ++i)
#pragma unroll
    for (int j = 0; j < 4; ++j)
#pragma unroll
      for (int e = 0; e < 8; ++e) acc[i][j][e] = 0.f;

  const int steps = K >> 5;

  // --- staging helpers ---
  auto issueA = [&](int k0, int buf) {
    async_copy_b128(&sA[buf][ar0][ac0], &A[(size_t)(bm0 + ar0) * K + k0 + ac0]);
    async_copy_b128(&sA[buf][ar1][ac1], &A[(size_t)(bm0 + ar1) * K + k0 + ac1]);
  };
  auto loadB = [&](int k0, float (&regs)[16]) {
#pragma unroll
    for (int i = 0; i < 16; ++i)
      regs[i] = __builtin_nontemporal_load(
          &Bw[(size_t)(k0 + bk0 + i) * N + bn0 + bn_]);
  };
  auto storeB = [&](int buf, const float (&regs)[16]) {
    v8bf lo, hi;
#pragma unroll
    for (int i = 0; i < 8; ++i) {
      lo[i] = (bf16_t)regs[i];
      hi[i] = (bf16_t)regs[8 + i];
    }
    *(v8bf*)&sB[buf][bn_][bk0]     = lo;   // ds_store_b128
    *(v8bf*)&sB[buf][bn_][bk0 + 8] = hi;   // ds_store_b128
  };

  // --- prologue: stage tile 0 ---
  {
    float b0[16];
    issueA(0, 0);
    loadB(0, b0);
    storeB(0, b0);
    wait_async0();
    __syncthreads();
  }

  for (int i = 0; i < steps; ++i) {
    const int buf = i & 1;
    const bool more = (i + 1 < steps);
    float bnext[16];
    if (more) {
      issueA((i + 1) << 5, buf ^ 1);   // async DMA overlaps compute
      loadB((i + 1) << 5, bnext);      // NT global loads overlap compute
    }
#pragma unroll
    for (int mt = 0; mt < 2; ++mt) {
      const int ar = wm * 32 + mt * 16 + l16;
      v16bf af = combine8(*(const v8bf*)&sA[buf][ar][half * 8],
                          *(const v8bf*)&sA[buf][ar][16 + half * 8]);
#pragma unroll
      for (int nt = 0; nt < 4; ++nt) {
        const int bc = wn * 64 + nt * 16 + l16;
        v16bf bf = combine8(*(const v8bf*)&sB[buf][bc][half * 8],
                            *(const v8bf*)&sB[buf][bc][16 + half * 8]);
        acc[mt][nt] = wmma_bf16(af, bf, acc[mt][nt]);
      }
    }
    if (more) {
      storeB(buf ^ 1, bnext);
      wait_async0();
      __syncthreads();
    }
  }

  // --- epilogue (residual decision hoisted; no per-element branches) ---
  const int row0 = bm0 + wm * 32 + half * 8;
  const int col0 = bn0 + wn * 64 + l16;
  if (R != nullptr) {
#pragma unroll
    for (int mt = 0; mt < 2; ++mt)
#pragma unroll
      for (int nt = 0; nt < 4; ++nt) {
        size_t off = (size_t)(row0 + mt * 16) * N + (col0 + nt * 16);
#pragma unroll
        for (int r = 0; r < 8; ++r, off += N)
          C[off] = acc[mt][nt][r] + R[off];
      }
  } else {
#pragma unroll
    for (int mt = 0; mt < 2; ++mt)
#pragma unroll
      for (int nt = 0; nt < 4; ++nt) {
        size_t off = (size_t)(row0 + mt * 16) * N + (col0 + nt * 16);
#pragma unroll
        for (int r = 0; r < 8; ++r, off += N)
          C[off] = acc[mt][nt][r];
      }
  }
}

// ---------------------------------------------------------------------------
// Fused SwiGLU gate GEMM: G[M,HID](bf16) = silu(A@W1) * (A@W3)
// Block tile 128x64, wave tile 32x32, double-buffered like above.
// Each thread owns a contiguous 8-wide k-run -> one ds_store_b128 per matrix.
// ---------------------------------------------------------------------------
__global__ __launch_bounds__(256)
void gemm_gate_kernel(const bf16_t* __restrict__ A, const float* __restrict__ W1,
                      const float* __restrict__ W3, bf16_t* __restrict__ G) {
  __shared__ bf16_t sA[2][128][32];   // 16 KB
  __shared__ bf16_t sB1[2][64][32];   //  8 KB
  __shared__ bf16_t sB3[2][64][32];   //  8 KB
  const int K = DIM_, N = HID_;
  const int tid  = threadIdx.x;
  const int lane = tid & 31, wave = tid >> 5;
  const int half = lane >> 4, l16 = lane & 15;
  const int wm = wave >> 1, wn = wave & 1;
  const int bm0 = blockIdx.y * 128, bn0 = blockIdx.x * 64;

  const int ar0 = tid >> 2, ac0 = (tid & 3) << 3;
  const int ar1 = (tid + 256) >> 2, ac1 = ((tid + 256) & 3) << 3;
  const int bn_ = tid & 63;            // B: column owned by this thread
  const int bk0 = (tid >> 6) << 3;     // contiguous k-run start (0,8,16,24)

  v8f acc1[2][2], acc3[2][2];
#pragma unroll
  for (int i = 0; i < 2; ++i)
#pragma unroll
    for (int j = 0; j < 2; ++j)
#pragma unroll
      for (int e = 0; e < 8; ++e) { acc1[i][j][e] = 0.f; acc3[i][j][e] = 0.f; }

  const int steps = K >> 5;   // 128

  auto issueA = [&](int k0, int buf) {
    async_copy_b128(&sA[buf][ar0][ac0], &A[(size_t)(bm0 + ar0) * K + k0 + ac0]);
    async_copy_b128(&sA[buf][ar1][ac1], &A[(size_t)(bm0 + ar1) * K + k0 + ac1]);
  };
  auto loadB = [&](int k0, float (&r1)[8], float (&r3)[8]) {
#pragma unroll
    for (int i = 0; i < 8; ++i) {
      size_t off = (size_t)(k0 + bk0 + i) * N + bn0 + bn_;
      r1[i] = __builtin_nontemporal_load(&W1[off]);
      r3[i] = __builtin_nontemporal_load(&W3[off]);
    }
  };
  auto storeB = [&](int buf, const float (&r1)[8], const float (&r3)[8]) {
    v8bf c1, c3;
#pragma unroll
    for (int i = 0; i < 8; ++i) {
      c1[i] = (bf16_t)r1[i];
      c3[i] = (bf16_t)r3[i];
    }
    *(v8bf*)&sB1[buf][bn_][bk0] = c1;   // ds_store_b128
    *(v8bf*)&sB3[buf][bn_][bk0] = c3;   // ds_store_b128
  };

  {
    float r1[8], r3[8];
    issueA(0, 0);
    loadB(0, r1, r3);
    storeB(0, r1, r3);
    wait_async0();
    __syncthreads();
  }

  for (int i = 0; i < steps; ++i) {
    const int buf = i & 1;
    const bool more = (i + 1 < steps);
    float r1n[8], r3n[8];
    if (more) {
      issueA((i + 1) << 5, buf ^ 1);
      loadB((i + 1) << 5, r1n, r3n);
    }
#pragma unroll
    for (int mt = 0; mt < 2; ++mt) {
      const int ar = wm * 32 + mt * 16 + l16;
      v16bf af = combine8(*(const v8bf*)&sA[buf][ar][half * 8],
                          *(const v8bf*)&sA[buf][ar][16 + half * 8]);
#pragma unroll
      for (int nt = 0; nt < 2; ++nt) {
        const int bc = wn * 32 + nt * 16 + l16;
        v16bf b1 = combine8(*(const v8bf*)&sB1[buf][bc][half * 8],
                            *(const v8bf*)&sB1[buf][bc][16 + half * 8]);
        v16bf b3 = combine8(*(const v8bf*)&sB3[buf][bc][half * 8],
                            *(const v8bf*)&sB3[buf][bc][16 + half * 8]);
        acc1[mt][nt] = wmma_bf16(af, b1, acc1[mt][nt]);
        acc3[mt][nt] = wmma_bf16(af, b3, acc3[mt][nt]);
      }
    }
    if (more) {
      storeB(buf ^ 1, r1n, r3n);
      wait_async0();
      __syncthreads();
    }
  }

  const int row0 = bm0 + wm * 32 + half * 8;
  const int col0 = bn0 + wn * 32 + l16;
#pragma unroll
  for (int mt = 0; mt < 2; ++mt)
#pragma unroll
    for (int nt = 0; nt < 2; ++nt) {
      size_t off = (size_t)(row0 + mt * 16) * N + (col0 + nt * 16);
#pragma unroll
      for (int r = 0; r < 8; ++r, off += N) {
        const float u = acc1[mt][nt][r];
        const float t = acc3[mt][nt][r];
        G[off] = (bf16_t)((u / (1.f + __expf(-u))) * t);   // silu(u) * t
      }
    }
}

// ---------------------------------------------------------------------------
// Causal GQA attention. One wave per (b, head, 16-query-row tile).
// Pass 1: scores via WMMA into LDS (bf16), causal-limited key range.
// Pass 2: exact per-row softmax (f32 math), probs stored bf16 in place.
// Pass 3: P @ V via WMMA with a transposed 128x32 V tile staged in LDS
// (V rows read as b128 vectors; scatter b16 stores do the transpose).
// ---------------------------------------------------------------------------
__global__ __launch_bounds__(32)
void attn_kernel(const bf16_t* __restrict__ Q, const bf16_t* __restrict__ Kc,
                 const bf16_t* __restrict__ Vc, bf16_t* __restrict__ O) {
  __shared__ bf16_t sS[16][S_];     // 32 KB: scores then probs
  __shared__ bf16_t sVt[HD_][32];   //  8 KB: transposed V chunk
  __shared__ float  sInv[16];

  const int lane = threadIdx.x;
  const int half = lane >> 4, l16 = lane & 15;
  const int qt = blockIdx.x, h = blockIdx.y, b = blockIdx.z;
  const int kvh = h / NREP_;
  const size_t qbase = (size_t)b * S_ * NH_ * HD_ + (size_t)h * HD_;
  const size_t kbase = (size_t)b * S_ * NKVH_ * HD_ + (size_t)kvh * HD_;
  const int qrow0 = qt * 16;
  const int nkeys = qrow0 + 16;                  // causal key count (tile end)
  const int nkt = qt + 1;                        // 16-wide key tiles
  const int kpad = ((nkeys + 31) >> 5) << 5;     // padded to 32 for PV
  const float scale = 0.08838834764831845f;      // 1/sqrt(128)

  // Q fragments for the whole HD=128 (4 chunks of K=32)
  v16bf qf[4];
#pragma unroll
  for (int c = 0; c < 4; ++c) {
    const bf16_t* qp = Q + qbase + (size_t)(qrow0 + l16) * NH_ * HD_ + c * 32;
    qf[c] = combine8(*(const v8bf*)&qp[half * 8],
                     *(const v8bf*)&qp[16 + half * 8]);
  }

  // ---- Pass 1: score tiles -> LDS (bf16) ----
  for (int kt = 0; kt < nkt; ++kt) {
    v8f cfr;
#pragma unroll
    for (int e = 0; e < 8; ++e) cfr[e] = 0.f;
    const bf16_t* kp = Kc + kbase + (size_t)(kt * 16 + l16) * NKVH_ * HD_;
#pragma unroll
    for (int cc = 0; cc < 4; ++cc) {
      v16bf kf = combine8(*(const v8bf*)&kp[cc * 32 + half * 8],
                          *(const v8bf*)&kp[cc * 32 + 16 + half * 8]);
      cfr = wmma_bf16(qf[cc], kf, cfr);
    }
    const int col = kt * 16 + l16;
#pragma unroll
    for (int r = 0; r < 8; ++r) {
      const int row = half * 8 + r;
      float s = cfr[r] * scale;
      if (col > qrow0 + row) s = -1e9f;          // causal mask
      sS[row][col] = (bf16_t)s;
    }
  }
  // zero-pad the [nkeys, kpad) prob columns
  for (int j = nkeys + lane; j < kpad; j += 32)
#pragma unroll
    for (int row = 0; row < 16; ++row) sS[row][j] = (bf16_t)0.f;
  __syncthreads();

  // ---- Pass 2: softmax, one row per lane (lanes 0..15) ----
  if (lane < 16) {
    const int row = lane;
    const int limit = qrow0 + row + 1;
    float m = -1e30f;
    for (int j = 0; j < limit; ++j) m = fmaxf(m, (float)sS[row][j]);
    float sum = 0.f;
    for (int j = 0; j < limit; ++j) {
      float p = __expf((float)sS[row][j] - m);
      sum += p;
      sS[row][j] = (bf16_t)p;
    }
    for (int j = limit; j < nkeys; ++j) sS[row][j] = (bf16_t)0.f;
    sInv[row] = 1.f / sum;
  }
  __syncthreads();

  // ---- Pass 3: P @ V ----
  v8f o[8];
#pragma unroll
  for (int nt = 0; nt < 8; ++nt)
#pragma unroll
    for (int e = 0; e < 8; ++e) o[nt][e] = 0.f;

  for (int k0 = 0; k0 < kpad; k0 += 32) {
    // stage transposed V chunk: lane owns key k0+lane; read b128, scatter b16
    const bf16_t* vp = Vc + kbase + (size_t)(k0 + lane) * NKVH_ * HD_;
#pragma unroll
    for (int d8 = 0; d8 < HD_ / 8; ++d8) {
      v8bf chunk = *(const v8bf*)&vp[d8 * 8];
#pragma unroll
      for (int e = 0; e < 8; ++e) sVt[d8 * 8 + e][lane] = chunk[e];
    }
    __syncthreads();
    v16bf pf = combine8(*(const v8bf*)&sS[l16][k0 + half * 8],
                        *(const v8bf*)&sS[l16][k0 + 16 + half * 8]);
#pragma unroll
    for (int nt = 0; nt < 8; ++nt) {
      v16bf vf = combine8(*(const v8bf*)&sVt[nt * 16 + l16][half * 8],
                          *(const v8bf*)&sVt[nt * 16 + l16][16 + half * 8]);
      o[nt] = wmma_bf16(pf, vf, o[nt]);
    }
    __syncthreads();
  }

  // epilogue: scale by 1/sum, write bf16
#pragma unroll
  for (int nt = 0; nt < 8; ++nt)
#pragma unroll
    for (int r = 0; r < 8; ++r) {
      const int row = half * 8 + r;
      const int col = nt * 16 + l16;
      O[qbase + (size_t)(qrow0 + row) * NH_ * HD_ + col] =
          (bf16_t)(o[nt][r] * sInv[row]);
    }
}

// ---------------------------------------------------------------------------
// Launch sequence
// ---------------------------------------------------------------------------
extern "C" void kernel_launch(void* const* d_in, const int* in_sizes, int n_in,
                              void* d_out, int out_size, void* d_ws, size_t ws_size,
                              hipStream_t stream) {
  (void)in_sizes; (void)n_in; (void)out_size; (void)ws_size;
  const float* x   = (const float*)d_in[0];
  const float* wq  = (const float*)d_in[1];
  const float* wk  = (const float*)d_in[2];
  const float* wv  = (const float*)d_in[3];
  const float* wo  = (const float*)d_in[4];
  const float* w1  = (const float*)d_in[5];
  const float* w2  = (const float*)d_in[6];
  const float* w3  = (const float*)d_in[7];
  const float* anw = (const float*)d_in[8];
  const float* fnw = (const float*)d_in[9];
  const float* fc  = (const float*)d_in[12];   // freqs_cis as float pairs
  float* out = (float*)d_out;

  // ---- workspace carving (~219 MB, 256B aligned regions) ----
  char* p = (char*)d_ws;
  auto carve = [&](size_t bytes) {
    char* r = p;
    p += (bytes + 255) & ~(size_t)255;
    return r;
  };
  bf16_t* hb = (bf16_t*)carve((size_t)M_ * DIM_ * 2);  // rmsnorm out (also fb)
  char* f32r = carve((size_t)M_ * DIM_ * 4 + 2 * (size_t)M_ * KVD_ * 4);
  float* xq = (float*)f32r;                 // qkv f32; region reused for g
  float* xk = xq + (size_t)M_ * DIM_;
  float* xv = xk + (size_t)M_ * KVD_;
  bf16_t* g  = (bf16_t*)f32r;               // [M, HID] bf16 (90 MB < 100 MB)
  bf16_t* qb = (bf16_t*)carve((size_t)M_ * DIM_ * 2);
  bf16_t* kb = (bf16_t*)carve((size_t)M_ * KVD_ * 2);
  bf16_t* vb = (bf16_t*)carve((size_t)M_ * KVD_ * 2);
  bf16_t* ab = (bf16_t*)carve((size_t)M_ * DIM_ * 2);

  // 1) attention RMSNorm
  rmsnorm_kernel<<<M_, 256, 0, stream>>>(x, anw, hb, DIM_);
  // 2) q/k/v projections (bf16 WMMA, fp32 out)
  gemm_bf16_kernel<<<dim3(DIM_ / 128, M_ / 128), 256, 0, stream>>>(hb, wq, xq, nullptr, DIM_, DIM_);
  gemm_bf16_kernel<<<dim3(KVD_ / 128, M_ / 128), 256, 0, stream>>>(hb, wk, xk, nullptr, KVD_, DIM_);
  gemm_bf16_kernel<<<dim3(KVD_ / 128, M_ / 128), 256, 0, stream>>>(hb, wv, xv, nullptr, KVD_, DIM_);
  // 3) RoPE + casts
  {
    size_t nq2 = (size_t)M_ * NH_ * (HD_ / 2);
    rope_cast_kernel<<<(unsigned)((nq2 + 255) / 256), 256, 0, stream>>>(xq, fc, qb, NH_, nq2);
    size_t nk2 = (size_t)M_ * NKVH_ * (HD_ / 2);
    rope_cast_kernel<<<(unsigned)((nk2 + 255) / 256), 256, 0, stream>>>(xk, fc, kb, NKVH_, nk2);
    size_t nv = (size_t)M_ * KVD_;
    cast_bf16_kernel<<<(unsigned)((nv + 255) / 256), 256, 0, stream>>>(xv, vb, nv);
  }
  // 4) causal GQA attention
  attn_kernel<<<dim3(S_ / 16, NH_, B_), 32, 0, stream>>>(qb, kb, vb, ab);
  // 5) output projection + residual x -> d_out (h_attn)
  gemm_bf16_kernel<<<dim3(DIM_ / 128, M_ / 128), 256, 0, stream>>>(ab, wo, out, x, DIM_, DIM_);
  // 6) FFN RMSNorm (reuses hb)
  rmsnorm_kernel<<<M_, 256, 0, stream>>>(out, fnw, hb, DIM_);
  // 7) fused silu(f@w1) * (f@w3) -> g (bf16)
  gemm_gate_kernel<<<dim3(HID_ / 64, M_ / 128), 256, 0, stream>>>(hb, w1, w3, g);
  // 8) g @ w2 + residual (in-place on d_out)
  gemm_bf16_kernel<<<dim3(DIM_ / 128, M_ / 128), 256, 0, stream>>>(g, w2, out, out, DIM_, HID_);
}
